// Model3_58274116272764
// MI455X (gfx1250) — matrix operand order repaired and checked by reading, not compiled, not run on hardware
//
#include <hip/hip_runtime.h>

// ---------------------------------------------------------------------------
// Types for CDNA5 WMMA (wave32, v_wmma_f32_16x16x32_f16)
// ---------------------------------------------------------------------------
typedef __attribute__((ext_vector_type(16))) _Float16     v16h;
typedef __attribute__((ext_vector_type(8)))  float        v8f;
typedef __attribute__((ext_vector_type(4)))  unsigned int v4u;

union HFrag { v16h v; v4u u[2]; };

__device__ __forceinline__ v8f wmma16(HFrag a, HFrag b, v8f c) {
  // 8 args: (neg_a, A, neg_b, B, c_mod, C, reuse_a, reuse_b)
  return __builtin_amdgcn_wmma_f32_16x16x32_f16(false, a.v, false, b.v,
                                                (short)0, c, false, false);
}

__device__ __forceinline__ void loadFrag(HFrag& f, const _Float16* p0, int delta) {
  f.u[0] = *reinterpret_cast<const v4u*>(p0);
  f.u[1] = *reinterpret_cast<const v4u*>(p0 + delta);
}

// ---------------------------------------------------------------------------
// Elementwise / conversion kernels
// ---------------------------------------------------------------------------
__global__ void k_cvt_f16(_Float16* dst, const float* src, long n) {
  long i = (long)blockIdx.x * blockDim.x + threadIdx.x;
  if (i < n) dst[i] = (_Float16)src[i];
}

// dst[c*rows + r] = src[r*cols + c]  (produce B[N,K] = W[K,N] transposed, f16)
__global__ void k_cvt_f16_T(_Float16* dst, const float* src, int rows, int cols) {
  long i = (long)blockIdx.x * blockDim.x + threadIdx.x;
  if (i >= (long)rows * cols) return;
  int r = (int)(i / cols), c = (int)(i % cols);
  dst[(long)c * rows + r] = (_Float16)src[i];
}

__global__ void k_gather_embed(_Float16* dst, const float* emb, const int* idx,
                               long ntok, int Dd) {
  long i = (long)blockIdx.x * blockDim.x + threadIdx.x;
  if (i >= ntok * (long)Dd) return;
  long tok = i / Dd;
  int  c   = (int)(i % Dd);
  int  w   = idx[tok];
  dst[i] = (_Float16)emb[(long)w * Dd + c];
}

__global__ void k_lens(int* lens, const int* x, int Ss, int Tt) {
  int s = blockIdx.x * blockDim.x + threadIdx.x;
  if (s >= Ss) return;
  int n = 0;
  for (int t = 0; t < Tt; ++t) n += (x[s * Tt + t] != 0) ? 1 : 0;
  lens[s] = n;
}

__global__ void k_init_state(float* Hf, _Float16* Hh, float* rmax, long n) {
  long i = (long)blockIdx.x * blockDim.x + threadIdx.x;
  if (i >= n) return;
  Hf[i] = 0.f;
  Hh[i] = (_Float16)0.f;
  rmax[i] = -3.0e38f;
}

// ---------------------------------------------------------------------------
// Fused GRU step GEMM, 32x64 output per wave (two 16-row subtiles share B).
// Rows r in [0, 2*Bpad): direction d = r/Bpad, batch element s = r - d*Bpad.
// A row = [ input(s, tt) (Din f16) | h_prev(r) (256 f16) ], tt = d? Ts-1-t : t
// Output G[r, 0:1024]:
//   cols [0,256)   : ir+hr   (K = Din+256)
//   cols [256,512) : iz+hz   (K = Din+256)
//   cols [512,768) : inn     (K = Din, input only;  Wih rows 512..767)
//   cols [768,1024): hn      (K = 256, hidden only; Whh rows 512..767)
// Wih: [2, 768, Din] f16,  Whh: [2, 768, 256] f16 (row-major == B[N,K])
// Out-of-batch A rows are CLAMPED (not branched); their C rows are masked
// at store, so the garbage never escapes.
// ---------------------------------------------------------------------------
__global__ void __launch_bounds__(128)
k_gru_step_gemm(const _Float16* __restrict__ in, const _Float16* __restrict__ Hh,
                const _Float16* __restrict__ Wih, const _Float16* __restrict__ Whh,
                float* __restrict__ G, int Bs, int Bpad, int Ts, int t, int Din) {
  int lane = threadIdx.x & 31;
  int mtd    = Bpad >> 4;            // 16-row tiles per direction
  int mpairs = (mtd + 1) >> 1;
  int wid = blockIdx.x * 4 + (threadIdx.x >> 5);
  if (wid >= 2 * mpairs * 16) return;
  int d   = wid / (mpairs * 16);
  int rem = wid - d * (mpairs * 16);
  int mp    = rem >> 4;
  int n0    = (rem & 15) << 6;
  int m0    = d * Bpad + (mp << 5);
  int m1    = m0 + 16;
  bool hasB = ((mp << 5) + 16) < Bpad;

  const _Float16* WihD = Wih + (size_t)d * 768 * Din;
  const _Float16* WhhD = Whh + (size_t)d * 768 * 256;

  int hlf = lane >> 4;
  int nB  = lane & 15;
  int koffB = hlf << 3;
  int tt = d ? (Ts - 1 - t) : t;

  int sr0 = (m0 + (lane & 15)) - d * Bpad; if (sr0 >= Bs) sr0 = Bs - 1;
  int sr1 = (m1 + (lane & 15)) - d * Bpad; if (sr1 >= Bs) sr1 = Bs - 1;

  const _Float16* aIn0 = in + ((size_t)sr0 * Ts + tt) * Din + hlf * 8;
  const _Float16* aIn1 = in + ((size_t)sr1 * Ts + tt) * Din + hlf * 8;
  const _Float16* aH0  = Hh + ((size_t)d * Bpad + sr0) * 256 + hlf * 8;
  const _Float16* aH1  = Hh + ((size_t)d * Bpad + sr1) * 256 + hlf * 8;

  v8f acc0[4] = {};
  v8f acc1[4] = {};

  // ---- phase 1: K over the input (tiles n0 < 768) ----
  if (n0 < 768) {
    const _Float16* b0 = WihD + (size_t)(n0 +  0 + nB) * Din + koffB;
    const _Float16* b1 = WihD + (size_t)(n0 + 16 + nB) * Din + koffB;
    const _Float16* b2 = WihD + (size_t)(n0 + 32 + nB) * Din + koffB;
    const _Float16* b3 = WihD + (size_t)(n0 + 48 + nB) * Din + koffB;
    const _Float16* a0 = aIn0;
    const _Float16* a1 = aIn1;
    for (int k0 = 0; k0 < Din; k0 += 32) {
      HFrag A0, A1, B0, B1, B2, B3;
      loadFrag(A0, a0, 16); loadFrag(A1, a1, 16);
      loadFrag(B0, b0, 16); loadFrag(B1, b1, 16);
      loadFrag(B2, b2, 16); loadFrag(B3, b3, 16);
      acc0[0] = wmma16(A0, B0, acc0[0]); acc1[0] = wmma16(A1, B0, acc1[0]);
      acc0[1] = wmma16(A0, B1, acc0[1]); acc1[1] = wmma16(A1, B1, acc1[1]);
      acc0[2] = wmma16(A0, B2, acc0[2]); acc1[2] = wmma16(A1, B2, acc1[2]);
      acc0[3] = wmma16(A0, B3, acc0[3]); acc1[3] = wmma16(A1, B3, acc1[3]);
      a0 += 32; a1 += 32; b0 += 32; b1 += 32; b2 += 32; b3 += 32;
    }
  }
  // ---- phase 2: K over the hidden state (tiles n0 < 512 or n0 >= 768) ----
  if (n0 < 512 || n0 >= 768) {
    int nw = (n0 < 512) ? n0 : (n0 - 256);   // hn uses Whh rows 512..767
    const _Float16* b0 = WhhD + (size_t)(nw +  0 + nB) * 256 + koffB;
    const _Float16* b1 = WhhD + (size_t)(nw + 16 + nB) * 256 + koffB;
    const _Float16* b2 = WhhD + (size_t)(nw + 32 + nB) * 256 + koffB;
    const _Float16* b3 = WhhD + (size_t)(nw + 48 + nB) * 256 + koffB;
    const _Float16* a0 = aH0;
    const _Float16* a1 = aH1;
    for (int k0 = 0; k0 < 256; k0 += 32) {
      HFrag A0, A1, B0, B1, B2, B3;
      loadFrag(A0, a0, 16); loadFrag(A1, a1, 16);
      loadFrag(B0, b0, 16); loadFrag(B1, b1, 16);
      loadFrag(B2, b2, 16); loadFrag(B3, b3, 16);
      acc0[0] = wmma16(A0, B0, acc0[0]); acc1[0] = wmma16(A1, B0, acc1[0]);
      acc0[1] = wmma16(A0, B1, acc0[1]); acc1[1] = wmma16(A1, B1, acc1[1]);
      acc0[2] = wmma16(A0, B2, acc0[2]); acc1[2] = wmma16(A1, B2, acc1[2]);
      acc0[3] = wmma16(A0, B3, acc0[3]); acc1[3] = wmma16(A1, B3, acc1[3]);
      a0 += 32; a1 += 32; b0 += 32; b1 += 32; b2 += 32; b3 += 32;
    }
  }

  // ---- store (C layout: vgpr v -> row v + 8*hlf, col = lane&15) ----
#pragma unroll
  for (int j = 0; j < 4; ++j) {
#pragma unroll
    for (int v = 0; v < 8; ++v) {
      int col = n0 + (j << 4) + nB;
      int r0 = m0 + v + (hlf << 3);
      if (r0 - d * Bpad < Bs) G[(size_t)r0 * 1024 + col] = acc0[j][v];
      int r1 = m1 + v + (hlf << 3);
      if (hasB && (r1 - d * Bpad < Bs)) G[(size_t)r1 * 1024 + col] = acc1[j][v];
    }
  }
}

// ---------------------------------------------------------------------------
// GRU gating + running max-pool over valid timesteps
// ---------------------------------------------------------------------------
__global__ void k_gru_gate(const float* __restrict__ G, float* __restrict__ Hf,
                           _Float16* __restrict__ Hh, float* __restrict__ rmax,
                           const float* __restrict__ bih, const float* __restrict__ bhh,
                           const int* __restrict__ lens, int Bs, int Bpad, int Ts, int t) {
  long i = (long)blockIdx.x * blockDim.x + threadIdx.x;
  long total = (long)2 * Bpad * 256;
  if (i >= total) return;
  int r = (int)(i >> 8);
  int c = (int)(i & 255);
  int d = r / Bpad;
  int s = r - d * Bpad;
  if (s >= Bs) return;
  const float* bi = bih + d * 768;
  const float* bh = bhh + d * 768;
  const float* g  = G + (size_t)r * 1024;
  float rp  = g[c]       + bi[c]       + bh[c];
  float zp  = g[256 + c] + bi[256 + c] + bh[256 + c];
  float inn = g[512 + c] + bi[512 + c];
  float hn  = g[768 + c] + bh[512 + c];
  float rg = 1.f / (1.f + __expf(-rp));
  float zg = 1.f / (1.f + __expf(-zp));
  float ng = tanhf(inn + rg * hn);
  float h  = Hf[i];
  float h2 = (1.f - zg) * ng + zg * h;
  Hf[i] = h2;
  Hh[i] = (_Float16)h2;
  int len = lens ? lens[s] : Ts;
  bool inc = d ? (t >= Ts - len) : (t < len);
  if (inc) rmax[i] = fmaxf(rmax[i], h2);
}

// concat fwd/bwd running max -> [Bs, 512] f32 + f16
__global__ void k_build_vec(float* vf, _Float16* vh, const float* rmax,
                            const int* lens, int Bs, int Bpad) {
  long i = (long)blockIdx.x * blockDim.x + threadIdx.x;
  if (i >= (long)Bs * 512) return;
  int s = (int)(i >> 9);
  int c = (int)(i & 511);
  float val = (c < 256) ? rmax[(size_t)s * 256 + c]
                        : rmax[((size_t)Bpad + s) * 256 + (c - 256)];
  if (lens && lens[s] == 0) val = 0.f;
  vf[i] = val;
  vh[i] = (_Float16)val;
}

// ---------------------------------------------------------------------------
// Generic WMMA GEMM: C[M,N] = A[M,K] * B[N,K]^T  (A,B f16, C f32), 32x64/wave
// ---------------------------------------------------------------------------
__global__ void __launch_bounds__(128)
k_gemm_f16(const _Float16* __restrict__ A, int lda,
           const _Float16* __restrict__ B, int ldb,
           float* __restrict__ C, int ldc, int M, int N, int K) {
  int lane = threadIdx.x & 31;
  int mt     = (M + 15) >> 4;
  int mpairs = (mt + 1) >> 1;
  int Nt = N >> 6;
  int wid = blockIdx.x * 4 + (threadIdx.x >> 5);
  if (wid >= mpairs * Nt) return;
  int mp = wid / Nt;
  int n0 = (wid - mp * Nt) << 6;
  int m0 = mp << 5;
  int m1 = m0 + 16;
  int hlf = lane >> 4, nB = lane & 15, koffB = hlf << 3;
  int r0 = m0 + (lane & 15); if (r0 >= M) r0 = M - 1;
  int r1 = m1 + (lane & 15); if (r1 >= M) r1 = M - 1;
  const _Float16* a0 = A + (size_t)r0 * lda + hlf * 8;
  const _Float16* a1 = A + (size_t)r1 * lda + hlf * 8;
  const _Float16* b0 = B + (size_t)(n0 +  0 + nB) * ldb + koffB;
  const _Float16* b1 = B + (size_t)(n0 + 16 + nB) * ldb + koffB;
  const _Float16* b2 = B + (size_t)(n0 + 32 + nB) * ldb + koffB;
  const _Float16* b3 = B + (size_t)(n0 + 48 + nB) * ldb + koffB;
  v8f acc0[4] = {};
  v8f acc1[4] = {};
  for (int k0 = 0; k0 < K; k0 += 32) {
    HFrag A0, A1, B0, B1, B2, B3;
    loadFrag(A0, a0, 16); loadFrag(A1, a1, 16);
    loadFrag(B0, b0, 16); loadFrag(B1, b1, 16);
    loadFrag(B2, b2, 16); loadFrag(B3, b3, 16);
    acc0[0] = wmma16(A0, B0, acc0[0]); acc1[0] = wmma16(A1, B0, acc1[0]);
    acc0[1] = wmma16(A0, B1, acc0[1]); acc1[1] = wmma16(A1, B1, acc1[1]);
    acc0[2] = wmma16(A0, B2, acc0[2]); acc1[2] = wmma16(A1, B2, acc1[2]);
    acc0[3] = wmma16(A0, B3, acc0[3]); acc1[3] = wmma16(A1, B3, acc1[3]);
    a0 += 32; a1 += 32; b0 += 32; b1 += 32; b2 += 32; b3 += 32;
  }
#pragma unroll
  for (int j = 0; j < 4; ++j) {
#pragma unroll
    for (int v = 0; v < 8; ++v) {
      int col = n0 + (j << 4) + nB;
      int w0 = m0 + v + (hlf << 3);
      if (w0 < M) C[(size_t)w0 * ldc + col] = acc0[j][v];
      int w1 = m1 + v + (hlf << 3);
      if (w1 < M) C[(size_t)w1 * ldc + col] = acc1[j][v];
    }
  }
}

// ---------------------------------------------------------------------------
// Scoring-head epilogues (tiny FLOPs; scalar)
// ---------------------------------------------------------------------------
__global__ void k_evsum(float* evsum, const float* evv, int En) {
  int c = blockIdx.x * blockDim.x + threadIdx.x;
  if (c >= 512) return;
  float a = 0.f;
  for (int e = 0; e < En; ++e) a += evv[(size_t)e * 512 + c];
  evsum[c] = a / (float)En;
}

__global__ void k_build_u(float* u, const float* W_ec, const float* W_esal,
                          const float* evsum) {
  int dd = blockIdx.x * blockDim.x + threadIdx.x;
  if (dd >= 512) return;
  const float* row = W_esal + (size_t)dd * 512;
  float a = 0.f;
  for (int k = 0; k < 512; ++k) a += row[k] * evsum[k];
  u[dd] = W_ec[dd] + a;
}

__global__ void k_event_probs(float* ep, const float* evv, const float* u,
                              const float* tfs, const float* tfp,
                              const float* bp, int En) {
  int e = blockIdx.x * blockDim.x + threadIdx.x;
  if (e >= En) return;
  const float* v = evv + (size_t)e * 512;
  float a = 0.f;
  for (int c = 0; c < 512; ++c) a += v[c] * u[c];
  ep[e] = a + tfp[0] * tfs[e] + bp[0];
}

__global__ void k_event_rel(float* erel, const float* q, const float* evv,
                            const float* ep, const float* b_erel,
                            const float* para, int Ss, int Kn) {
  int s = blockIdx.x * blockDim.x + threadIdx.x;
  if (s >= Ss) return;
  const float* qs = q + (size_t)s * 512;
  float acc = 0.f;
  for (int j = 0; j < Kn; ++j) {
    const float* v = evv + ((size_t)s * Kn + j) * 512;
    float sim = 0.f;
    for (int c = 0; c < 512; ++c) sim += qs[c] * v[c];
    acc += (sim + b_erel[0]) * ep[s * Kn + j];
  }
  erel[s] = para[0] * acc;
}

__global__ void k_ctx(float* ctx, const float* blog, const float* docv, int NDn) {
  long i = (long)blockIdx.x * blockDim.x + threadIdx.x;
  if (i >= (long)NDn * 1024) return;
  int n = (int)(i >> 10);
  int c = (int)(i & 1023);
  ctx[i] = (c < 512) ? blog[c] : docv[(size_t)n * 512 + (c - 512)];
}

__global__ void k_sent_probs(float* sp, const float* sv, const float* t2,
                             const float* ctx, const float* W_sc,
                             const float* dpe, const float* spe,
                             const float* W_sdp, const float* W_sp,
                             const float* erel, const float* biasp,
                             int Ss, int Lc, int NDn) {
  int s = blockIdx.x * blockDim.x + threadIdx.x;
  if (s >= Ss) return;
  const float* v = sv + (size_t)s * 512;
  float a = 0.f;
  for (int c = 0; c < 512; ++c) a += v[c] * W_sc[c];
  int n = s / Lc;
  int l = s - n * Lc;
  const float* t  = t2 + (size_t)s * 1024;
  const float* cx = ctx + (size_t)n * 1024;
  float b = 0.f;
  for (int c = 0; c < 1024; ++c) b += t[c] * cx[c];
  int di = (n * 50) / NDn;   // PDOC = 50
  int si = (l * 50) / Lc;    // PSENT = 50
  float c1 = 0.f, c2 = 0.f;
  for (int p = 0; p < 64; ++p) c1 += dpe[di * 64 + p] * W_sdp[p];
  for (int p = 0; p < 64; ++p) c2 += spe[si * 64 + p] * W_sp[p];
  sp[s] = a + b + c1 + c2 + erel[s] + biasp[0];
}

// ---------------------------------------------------------------------------
// Host driver
// ---------------------------------------------------------------------------
extern "C" void kernel_launch(void* const* d_in, const int* in_sizes, int n_in,
                              void* d_out, int out_size, void* d_ws, size_t ws_size,
                              hipStream_t stream) {
  (void)in_sizes; (void)n_in; (void)out_size; (void)ws_size;
  const int S = 2000, T = 50, D = 256, E = 4000, EW = 4, ND = 100, Lc = 20, Kn = 2;

  const int*   x        = (const int*)d_in[0];
  const int*   events   = (const int*)d_in[1];
  const float* evtfs    = (const float*)d_in[5];
  const float* embed    = (const float*)d_in[8];
  const float* wWih = (const float*)d_in[9],  *wWhh = (const float*)d_in[10];
  const float* wbih = (const float*)d_in[11], *wbhh = (const float*)d_in[12];
  const float* sWih = (const float*)d_in[13], *sWhh = (const float*)d_in[14];
  const float* sbih = (const float*)d_in[15], *sbhh = (const float*)d_in[16];
  const float* dWih = (const float*)d_in[17], *dWhh = (const float*)d_in[18];
  const float* dbih = (const float*)d_in[19], *dbhh = (const float*)d_in[20];
  const float* eWih = (const float*)d_in[21], *eWhh = (const float*)d_in[22];
  const float* ebih = (const float*)d_in[23], *ebhh = (const float*)d_in[24];
  const float* dpe  = (const float*)d_in[25], *spe  = (const float*)d_in[26];
  const float* W_ec   = (const float*)d_in[27];
  const float* W_esal = (const float*)d_in[28];
  const float* W_sc   = (const float*)d_in[29];
  const float* W_ssal = (const float*)d_in[30];
  const float* W_sdp  = (const float*)d_in[31];
  const float* W_sp   = (const float*)d_in[32];
  const float* W_erel = (const float*)d_in[33];
  const float* b_erel = (const float*)d_in[34];
  const float* tf_p   = (const float*)d_in[35];
  const float* ebias_p = (const float*)d_in[36];
  const float* para_p  = (const float*)d_in[37];
  const float* sbias_p = (const float*)d_in[39];

  float* out_sp = (float*)d_out;      // sent_probs [2000]
  float* out_ep = (float*)d_out + S;  // event_probs [4000]

  // workspace bump allocator
  char* base = (char*)d_ws;
  size_t off = 0;
  auto alloc = [&](size_t bytes) -> void* {
    void* p = base + off;
    off = (off + bytes + 255) & ~(size_t)255;
    return p;
  };
  _Float16* xe     = (_Float16*)alloc((size_t)S * T * D * 2);
  _Float16* evxe   = (_Float16*)alloc((size_t)E * EW * D * 2);
  _Float16* wWihH  = (_Float16*)alloc((size_t)2 * 768 * 256 * 2);
  _Float16* wWhhH  = (_Float16*)alloc((size_t)2 * 768 * 256 * 2);
  _Float16* eWihH  = (_Float16*)alloc((size_t)2 * 768 * 256 * 2);
  _Float16* eWhhH  = (_Float16*)alloc((size_t)2 * 768 * 256 * 2);
  _Float16* sWihH  = (_Float16*)alloc((size_t)2 * 768 * 512 * 2);
  _Float16* sWhhH  = (_Float16*)alloc((size_t)2 * 768 * 256 * 2);
  _Float16* dWihH  = (_Float16*)alloc((size_t)2 * 768 * 512 * 2);
  _Float16* dWhhH  = (_Float16*)alloc((size_t)2 * 768 * 256 * 2);
  _Float16* WerelT = (_Float16*)alloc((size_t)512 * 512 * 2);
  _Float16* WssalT = (_Float16*)alloc((size_t)1024 * 512 * 2);
  float*    Hf     = (float*)alloc((size_t)8000 * 256 * 4);
  _Float16* HhS    = (_Float16*)alloc((size_t)8000 * 256 * 2);
  float*    Gbuf   = (float*)alloc((size_t)8000 * 1024 * 4);
  float*    rmax   = (float*)alloc((size_t)8000 * 256 * 4);
  float*    sentv  = (float*)alloc((size_t)S * 512 * 4);
  _Float16* sentvH = (_Float16*)alloc((size_t)S * 512 * 2);
  float*    evv    = (float*)alloc((size_t)E * 512 * 4);
  _Float16* evvH   = (_Float16*)alloc((size_t)E * 512 * 2);
  float*    docv   = (float*)alloc((size_t)ND * 512 * 4);
  _Float16* docvH  = (_Float16*)alloc((size_t)ND * 512 * 2);
  float*    blog   = (float*)alloc((size_t)512 * 4);
  _Float16* blogH  = (_Float16*)alloc((size_t)512 * 2);
  float*    ctx    = (float*)alloc((size_t)ND * 1024 * 4);
  float*    evsum  = (float*)alloc(512 * 4);
  float*    uvec   = (float*)alloc(512 * 4);
  float*    qbuf   = (float*)alloc((size_t)S * 512 * 4);
  float*    t2buf  = (float*)alloc((size_t)S * 1024 * 4);
  float*    erel   = (float*)alloc((size_t)S * 4);
  int*      lens   = (int*)alloc((size_t)S * 4);

  auto nb = [](long n) { return (unsigned)((n + 255) / 256); };

  // --- f16 conversions + gathers ---
  k_cvt_f16<<<nb(2L*768*256), 256, 0, stream>>>(wWihH, wWih, 2L*768*256);
  k_cvt_f16<<<nb(2L*768*256), 256, 0, stream>>>(wWhhH, wWhh, 2L*768*256);
  k_cvt_f16<<<nb(2L*768*256), 256, 0, stream>>>(eWihH, eWih, 2L*768*256);
  k_cvt_f16<<<nb(2L*768*256), 256, 0, stream>>>(eWhhH, eWhh, 2L*768*256);
  k_cvt_f16<<<nb(2L*768*512), 256, 0, stream>>>(sWihH, sWih, 2L*768*512);
  k_cvt_f16<<<nb(2L*768*256), 256, 0, stream>>>(sWhhH, sWhh, 2L*768*256);
  k_cvt_f16<<<nb(2L*768*512), 256, 0, stream>>>(dWihH, dWih, 2L*768*512);
  k_cvt_f16<<<nb(2L*768*256), 256, 0, stream>>>(dWhhH, dWhh, 2L*768*256);
  k_cvt_f16_T<<<nb(512L*512),  256, 0, stream>>>(WerelT, W_erel, 512, 512);
  k_cvt_f16_T<<<nb(512L*1024), 256, 0, stream>>>(WssalT, W_ssal, 512, 1024);
  k_gather_embed<<<nb((long)S*T*D), 256, 0, stream>>>(xe, embed, x, (long)S*T, D);
  k_gather_embed<<<nb((long)E*EW*D), 256, 0, stream>>>(evxe, embed, events, (long)E*EW, D);
  k_lens<<<nb(S), 256, 0, stream>>>(lens, x, S, T);

  // --- generic bi-GRU runner (both directions batched, padded per dir) ---
  auto run_gru = [&](const _Float16* in, int Bs, int Bpad, int Ts, int Din,
                     const _Float16* WihH, const _Float16* WhhH,
                     const float* bih, const float* bhh, const int* lensp) {
    long nstate = (long)2 * Bpad * 256;
    k_init_state<<<nb(nstate), 256, 0, stream>>>(Hf, HhS, rmax, nstate);
    int mtd = Bpad >> 4;
    int mpairs = (mtd + 1) >> 1;
    int waves = 2 * mpairs * 16;
    unsigned gblocks = (unsigned)((waves + 3) / 4);
    for (int t = 0; t < Ts; ++t) {
      k_gru_step_gemm<<<gblocks, 128, 0, stream>>>(in, HhS, WihH, WhhH, Gbuf,
                                                   Bs, Bpad, Ts, t, Din);
      k_gru_gate<<<nb(nstate), 256, 0, stream>>>(Gbuf, Hf, HhS, rmax, bih, bhh,
                                                 lensp, Bs, Bpad, Ts, t);
    }
  };

  // word level -> sent_vec [S,512]
  run_gru(xe, S, S, T, 256, wWihH, wWhhH, wbih, wbhh, lens);
  k_build_vec<<<nb((long)S*512), 256, 0, stream>>>(sentv, sentvH, rmax, lens, S, S);

  // event level -> event_vec [E,512]
  run_gru(evxe, E, E, EW, 256, eWihH, eWhhH, ebih, ebhh, nullptr);
  k_build_vec<<<nb((long)E*512), 256, 0, stream>>>(evv, evvH, rmax, nullptr, E, E);

  // sentence level (docs of 20 sentences) -> doc_vec [ND,512]
  run_gru(sentvH, ND, 112, Lc, 512, sWihH, sWhhH, sbih, sbhh, nullptr);
  k_build_vec<<<nb((long)ND*512), 256, 0, stream>>>(docv, docvH, rmax, nullptr, ND, 112);

  // doc level (one sequence of 100 docs) -> blog_vec [512]
  run_gru(docvH, 1, 16, ND, 512, dWihH, dWhhH, dbih, dbhh, nullptr);
  k_build_vec<<<nb(512L), 256, 0, stream>>>(blog, blogH, rmax, nullptr, 1, 16);

  // --- event scoring ---
  k_evsum<<<2, 256, 0, stream>>>(evsum, evv, E);
  k_build_u<<<2, 256, 0, stream>>>(uvec, W_ec, W_esal, evsum);
  k_event_probs<<<nb(E), 256, 0, stream>>>(out_ep, evv, uvec, evtfs, tf_p, ebias_p, E);

  // q = sent_vec @ W_erel  (WMMA GEMM, B pre-transposed)
  {
    int mpairs = (((S + 15) / 16) + 1) / 2, Nt = 512 / 64;
    k_gemm_f16<<<(unsigned)((mpairs * Nt + 3) / 4), 128, 0, stream>>>(
        sentvH, 512, WerelT, 512, qbuf, 512, S, 512, 512);
  }
  k_event_rel<<<nb(S), 256, 0, stream>>>(erel, qbuf, evv, out_ep, b_erel, para_p, S, Kn);

  // --- sentence scoring ---
  k_ctx<<<nb((long)ND*1024), 256, 0, stream>>>(ctx, blog, docv, ND);
  {
    int mpairs = (((S + 15) / 16) + 1) / 2, Nt = 1024 / 64;
    k_gemm_f16<<<(unsigned)((mpairs * Nt + 3) / 4), 128, 0, stream>>>(
        sentvH, 512, WssalT, 512, t2buf, 1024, S, 1024, 512);
  }
  k_sent_probs<<<nb(S), 256, 0, stream>>>(out_sp, sentv, t2buf, ctx, W_sc, dpe, spe,
                                          W_sdp, W_sp, erel, sbias_p, S, Lc, ND);
}
